// RelationAttention_33054068310777
// MI455X (gfx1250) — compile-verified, matrix operand order checked by reading, not run on hardware
//
#include <hip/hip_runtime.h>
#include <cmath>

// Problem constants (from reference): B=8, S=512, H=256, A=64
#define BDIM 8
#define SDIM 512
#define HDIM 256
#define ADIM 64
#define MROWS (BDIM * SDIM)   // 4096 rows for the fused GEMMs

typedef __attribute__((ext_vector_type(16))) __bf16 v16bf;
typedef __attribute__((ext_vector_type(8)))  float  v8f;

// ---------------------------------------------------------------------------
// Fragment loader: 16-bit A/B WMMA layout (16x32 tile, K along H).
// Lane l (<16) holds row/col (base + l), K = k0+0..7 (elems 0..7) and
// K = k0+16..23 (elems 8..15). Lane l+16 holds K = k0+8..15 / k0+24..31.
// `p` must point at rowbase + k0 + hoff (hoff = 0 or 8), 16B-aligned.
// ---------------------------------------------------------------------------
__device__ __forceinline__ v16bf load_frag_bf16(const float* __restrict__ p) {
  float4 x0 = *(const float4*)(p + 0);
  float4 x1 = *(const float4*)(p + 4);
  float4 y0 = *(const float4*)(p + 16);
  float4 y1 = *(const float4*)(p + 20);
  v16bf f;
  f[0]  = (__bf16)x0.x; f[1]  = (__bf16)x0.y; f[2]  = (__bf16)x0.z; f[3]  = (__bf16)x0.w;
  f[4]  = (__bf16)x1.x; f[5]  = (__bf16)x1.y; f[6]  = (__bf16)x1.z; f[7]  = (__bf16)x1.w;
  f[8]  = (__bf16)y0.x; f[9]  = (__bf16)y0.y; f[10] = (__bf16)y0.z; f[11] = (__bf16)y0.w;
  f[12] = (__bf16)y1.x; f[13] = (__bf16)y1.y; f[14] = (__bf16)y1.z; f[15] = (__bf16)y1.w;
  return f;
}

// ---------------------------------------------------------------------------
// Kernel 1: fused dual GEMM via WMMA.
//   TA[row][a]   = RH(row,:) . Wta(a,:) + bias[a]   (bias folded here)
//   JAT[b][a][s] = RH(row,:) . Wja(a,:)             (stored transposed)
// One wave per 16x16 output tile; 256 M-tiles x 4 N-tiles = 1024 tiles.
// ---------------------------------------------------------------------------
__global__ __launch_bounds__(256) void k1_dual_gemm_wmma(
    const float* __restrict__ RH,   // (4096, 256)
    const float* __restrict__ Wta,  // (64, 256)
    const float* __restrict__ Wja,  // (64, 256)
    const float* __restrict__ bias, // (64)
    float* __restrict__ TA,         // (4096, 64)
    float* __restrict__ JAT) {      // (8, 64, 512)
  const int lane = threadIdx.x & 31;
  const int wave = threadIdx.x >> 5;
  const int tile = blockIdx.x * 8 + wave;      // 128 blocks * 8 waves = 1024
  const int mt = tile >> 2;                    // 0..255
  const int nt = tile & 3;                     // 0..3
  const int mbase = mt * 16;
  const int nbase = nt * 16;
  const int l15  = lane & 15;
  const int hoff = (lane < 16) ? 0 : 8;

  const float* arow = RH  + (size_t)(mbase + l15) * HDIM + hoff;
  const float* trow = Wta + (size_t)(nbase + l15) * HDIM + hoff;
  const float* jrow = Wja + (size_t)(nbase + l15) * HDIM + hoff;

  v8f accT = {};
  v8f accJ = {};
#pragma unroll 2
  for (int k0 = 0; k0 < HDIM; k0 += 32) {
    v16bf af = load_frag_bf16(arow + k0);
    v16bf bt = load_frag_bf16(trow + k0);
    v16bf bj = load_frag_bf16(jrow + k0);
    accT = __builtin_amdgcn_wmma_f32_16x16x32_bf16(false, af, false, bt,
                                                   (short)0, accT, false, false);
    accJ = __builtin_amdgcn_wmma_f32_16x16x32_bf16(false, af, false, bj,
                                                   (short)0, accJ, false, false);
  }

  // C/D layout: VGPR r, lanes 0-15 -> (M = mbase+r,   N = lane),
  //             lanes 16-31        -> (M = mbase+8+r, N = lane-16)
  const int col  = nbase + l15;
  const int radd = (lane < 16) ? 0 : 8;
  const float bcol = bias[col];
#pragma unroll
  for (int r = 0; r < 8; ++r) {
    const int row = mbase + radd + r;           // row = b*SDIM + s
    TA[(size_t)row * ADIM + col] = accT[r] + bcol;
    const int bb = row >> 9;                    // / SDIM
    const int s  = row & (SDIM - 1);
    JAT[(((size_t)bb * ADIM + col) << 9) + s] = accJ[r];
  }
}

// ---------------------------------------------------------------------------
// gfx1250 TRANS ops. __builtin_amdgcn_tanhf is called UNGUARDED on purpose:
// compile success proves the native v_tanh_f32 path lowered (the whole
// problem is tanh-throughput bound, so the OCML polynomial fallback would be
// the single biggest perf loss available here).
// ---------------------------------------------------------------------------
__device__ __forceinline__ float fast_tanh(float x) {
  return __builtin_amdgcn_tanhf(x);                        // v_tanh_f32
}

__device__ __forceinline__ float fast_exp(float x) {
  return __builtin_amdgcn_exp2f(x * 1.4426950408889634f);  // v_exp_f32
}

// ---------------------------------------------------------------------------
// Kernel 2: fused tanh-contraction + row softmax.
// One workgroup (512 threads = 16 waves) per (b, i) row; thread t owns j = t.
//   score[j] = sum_a v[a] * tanh(TA[b,i,a] + JAT[b,a,j])
//   out[b,i,:] = softmax(score)
// Reductions: 5-step __shfl_xor butterfly inside each wave32 (no barrier),
// then a 16-entry cross-wave LDS stage. Only 2 s_barriers per row.
// ---------------------------------------------------------------------------
__global__ __launch_bounds__(512) void k2_tanh_softmax(
    const float* __restrict__ TA,   // (4096, 64), bias folded in
    const float* __restrict__ JAT,  // (8, 64, 512)
    const float* __restrict__ v,    // (64)
    float* __restrict__ out) {      // (8, 512, 512)
  __shared__ float sTA[ADIM];
  __shared__ float sV[ADIM];
  __shared__ float sMax[16];
  __shared__ float sSum[16];

  const int bi   = blockIdx.x;      // b*SDIM + i
  const int b    = bi >> 9;
  const int j    = threadIdx.x;     // 0..511
  const int lane = j & 31;
  const int wv   = j >> 5;          // 0..15

  if (j < ADIM) {
    sTA[j] = TA[(size_t)bi * ADIM + j];
  } else if (j < 2 * ADIM) {
    sV[j - ADIM] = v[j - ADIM];
  }
  __syncthreads();

  const float* jat = JAT + ((size_t)b * ADIM << 9) + j;
  float acc = 0.0f;
#pragma unroll 8
  for (int a = 0; a < ADIM; ++a) {
    const float x = sTA[a] + jat[(size_t)a << 9];
    acc = fmaf(sV[a], fast_tanh(x), acc);
  }

  // --- softmax over the 512 j's in this workgroup ---
  // wave-local max (butterfly, wave32)
  float r = acc;
#pragma unroll
  for (int m = 16; m >= 1; m >>= 1) r = fmaxf(r, __shfl_xor(r, m, 32));
  if (lane == 0) sMax[wv] = r;
  __syncthreads();

  float rowmax = sMax[0];
#pragma unroll
  for (int w = 1; w < 16; ++w) rowmax = fmaxf(rowmax, sMax[w]);  // LDS broadcast reads

  const float e = fast_exp(acc - rowmax);

  // wave-local sum (butterfly)
  float s = e;
#pragma unroll
  for (int m = 16; m >= 1; m >>= 1) s += __shfl_xor(s, m, 32);
  if (lane == 0) sSum[wv] = s;      // distinct array from sMax -> no WAR barrier
  __syncthreads();

  float total = sSum[0];
#pragma unroll
  for (int w = 1; w < 16; ++w) total += sSum[w];

  const float inv = __builtin_amdgcn_rcpf(total);
  out[((size_t)bi << 9) + j] = e * inv;
}

// ---------------------------------------------------------------------------
// Launch. Inputs (setup_inputs order): relation_hidden, w_ta, w_ja, b, v.
// Workspace layout: TA (1 MB) | JAT (1 MB).
// ---------------------------------------------------------------------------
extern "C" void kernel_launch(void* const* d_in, const int* in_sizes, int n_in,
                              void* d_out, int out_size, void* d_ws, size_t ws_size,
                              hipStream_t stream) {
  (void)in_sizes; (void)n_in; (void)out_size; (void)ws_size;
  const float* RH   = (const float*)d_in[0];
  const float* Wta  = (const float*)d_in[1];
  const float* Wja  = (const float*)d_in[2];
  const float* bias = (const float*)d_in[3];
  const float* vvec = (const float*)d_in[4];
  float* out = (float*)d_out;

  float* TA  = (float*)d_ws;                       // 4096*64 floats = 1 MB
  float* JAT = TA + (size_t)MROWS * ADIM;          // 8*64*512 floats = 1 MB

  // 1024 tiles / 8 waves per block = 128 blocks
  hipLaunchKernelGGL(k1_dual_gemm_wmma, dim3(128), dim3(256), 0, stream,
                     RH, Wta, Wja, bias, TA, JAT);
  // one block per (b, i) row
  hipLaunchKernelGGL(k2_tanh_softmax, dim3(BDIM * SDIM), dim3(512), 0, stream,
                     TA, JAT, vvec, out);
}
